// Propgate_time_step_37520834298540
// MI455X (gfx1250) — matrix operand order repaired and checked by reading, not compiled, not run on hardware
//
#include <hip/hip_runtime.h>
#include <stdint.h>

#define HID 256
#define WHAT 50
#define WHERE 3
#define TSTEPS 16
#define NROW 4096
#define MT 16
#define NTHREADS 256
#define NWAVES 8

typedef unsigned short u16;
typedef __attribute__((ext_vector_type(8)))  __bf16 bf8v;
typedef __attribute__((ext_vector_type(16))) __bf16 bf16v;
typedef __attribute__((ext_vector_type(8)))  float  f8v;

// padded K / N dims: K multiples of 32; N padded to multiples of 128 (8 waves * 16)
#define K_REL 480    // rel_in 462
#define K_TEM 640    // tem_in 615
#define K_MT  576    // lwhat in 562
#define K_GLM 416    // glimpse 400
#define N_GE1P 256   // 200 -> 256 (16 tiles, NJ=2)
#define K_GE2 224    // 200
#define N_GE2P 128   // 100 -> 128 (8 tiles, NJ=1)
#define N_LWHATP 128 // 100 -> 128 (8 tiles, NJ=1)
#define ENC_W 112    // encbf LDS row stride
#define MT_W 112     // mt_raw LDS row stride

// bf16 weight workspace offsets (u16 elements)
#define OFF_WIH_REL 0u
#define OFF_WHH_REL (OFF_WIH_REL + 1024u*K_REL)
#define OFF_WIH_TEM (OFF_WHH_REL + 1024u*256u)
#define OFF_WHH_TEM (OFF_WIH_TEM + 1024u*K_TEM)
#define OFF_WGE1    (OFF_WHH_TEM + 1024u*256u)
#define OFF_WGE2    (OFF_WGE1 + (unsigned)(N_GE1P*K_GLM))
#define OFF_WLWHAT  (OFF_WGE2 + (unsigned)(N_GE2P*K_GE2))

// output offsets (floats)
#define OUT_ZWHAT   0u
#define OUT_ZWHERE  (OUT_ZWHAT  + (unsigned)(NROW*TSTEPS*WHAT))
#define OUT_ZPRES   (OUT_ZWHERE + (unsigned)(NROW*TSTEPS*WHERE))
#define OUT_KLWHAT  (OUT_ZPRES  + (unsigned)(NROW*TSTEPS))
#define OUT_KLWHERE (OUT_KLWHAT + (unsigned)NROW)
#define OUT_HTEMP   (OUT_KLWHERE + (unsigned)NROW)

__device__ __forceinline__ u16 f2bf(float x) {
  union { float f; unsigned int u; } v; v.f = x;
  unsigned int r = v.u + 0x7FFFu + ((v.u >> 16) & 1u);   // RNE
  return (u16)(r >> 16);
}
__device__ __forceinline__ float bf2f(u16 u) {
  union { unsigned int i; float f; } v; v.i = ((unsigned int)u) << 16; return v.f;
}
__device__ __forceinline__ float sigf(float x) { return 1.0f / (1.0f + expf(-x)); }
__device__ __forceinline__ float splus(float x) { return (x > 20.0f) ? x : log1pf(expf(x)); }

// ---------------------------------------------------------------------------
// Wave-level WMMA GEMM, guard-free: wave w owns output tiles t = w + 8*j,
// j = 0..NJ-1; B must be padded to NJ*128 rows. All NJ B fragments + the A
// fragment are loaded first (one load clause), then NJ back-to-back WMMAs.
// A layout per ISA: lanes0-15 K{0..7,16..23}, lanes16-31 K{8..15,24..31} (M=lane&15)
// B layout: lane covers N=lane&15; same K order as A (lanes0-15 K{0..7,16..23}, lanes16-31 K{8..15,24..31})
// ---------------------------------------------------------------------------
template<int NJ>
__device__ __forceinline__ void wmma_mm(const u16* A, int lda,
                                        const u16* __restrict__ B, int ldb,
                                        int kChunks, int wave, int lane, f8v acc[NJ]) {
  const int rA = lane & 15;
  const int hi = lane >> 4;
  const int aoff = hi ? 8 : 0;
  const int boff = hi ? 8 : 0;
  const u16* arow = A + rA * lda + aoff;
  const u16* brow = B + (size_t)(wave * 16 + rA) * ldb + boff;
  union BF { bf16v v; bf8v h[2]; };
  for (int kk = 0; kk < kChunks; ++kk) {
    const int kb = kk * 32;
    BF bfv[NJ];
#pragma unroll
    for (int j = 0; j < NJ; ++j) {
      const u16* bp = brow + (size_t)(128 * j) * ldb + kb;
      bfv[j].h[0] = *(const bf8v*)(bp);
      bfv[j].h[1] = *(const bf8v*)(bp + 16);
    }
    BF af;
    af.h[0] = *(const bf8v*)(arow + kb);
    af.h[1] = *(const bf8v*)(arow + kb + 16);
#pragma unroll
    for (int j = 0; j < NJ; ++j)
      acc[j] = __builtin_amdgcn_wmma_f32_16x16x32_bf16(
          false, af.v, false, bfv[j].v, (short)0, acc[j], false, false);
  }
}

struct KParams {
  const float *img, *zwt_last, *zwr_last, *zp_last, *hid_last;
  const float *eps_where, *eps_what, *u_pres;
  const float *w_pro_loca, *b_pro_loca, *b_ge1, *b_ge2;
  const float *bih_rel, *bhh_rel, *bih_tem, *bhh_tem;
  const float *w_lwhere, *b_lwhere, *b_lwhat, *w_lpres, *b_lpres;
  const u16 *W;
  float *out;
};

// LSTM gate epilogue: wave w owns unit blocks b0=w (j even) and b1=w+8 (j odd),
// gate g = j/2 (0=i,1=f,2=g,3=o). Gate col = 256*g + 16*b + (lane&15).
__device__ __forceinline__ void lstm_epilogue(const f8v* acc,
                                              const float* __restrict__ bih,
                                              const float* __restrict__ bhh,
                                              f8v* cAp, f8v* cBp, bool updC,
                                              u16* hOut, int wave, int lane) {
  const int rA = lane & 15, hi = lane >> 4;
  float bias[8];
#pragma unroll
  for (int j = 0; j < 8; ++j) {
    const int tt = wave + 8 * j;
    const int col = (j >> 1) * 256 + (tt & 15) * 16 + rA;
    bias[j] = bih[col] + bhh[col];
  }
  const int uA = wave * 16 + rA;
  const int uB = (wave + 8) * 16 + rA;
#pragma unroll
  for (int v = 0; v < 8; ++v) {
    const int m = v + hi * 8;
    {
      float i_ = sigf(acc[0][v] + bias[0]);
      float f_ = sigf(acc[2][v] + bias[2]);
      float g_ = tanhf(acc[4][v] + bias[4]);
      float o_ = sigf(acc[6][v] + bias[6]);
      float c2 = f_ * (*cAp)[v] + i_ * g_;
      if (updC) (*cAp)[v] = c2;
      hOut[m * HID + uA] = f2bf(o_ * tanhf(c2));
    }
    {
      float i_ = sigf(acc[1][v] + bias[1]);
      float f_ = sigf(acc[3][v] + bias[3]);
      float g_ = tanhf(acc[5][v] + bias[5]);
      float o_ = sigf(acc[7][v] + bias[7]);
      float c2 = f_ * (*cBp)[v] + i_ * g_;
      if (updC) (*cBp)[v] = c2;
      hOut[m * HID + uB] = f2bf(o_ * tanhf(c2));
    }
  }
}

// glimpse + 2-layer MLP encoder -> encbf (bf16 16x112, cols<100 valid)
// U: 10240 u16 scratch: g tile [0,6656), h1 tile [6656,10240)
__device__ __forceinline__ void glimpse_encode(const KParams& p, int row0,
                                               const float* zwLds, u16* U,
                                               u16* encbf, int tid, int wave, int lane) {
  for (int i = tid; i < 10240; i += NTHREADS) U[i] = 0;
  __syncthreads();
  for (int idx = tid; idx < MT * 400; idx += NTHREADS) {
    const int r = idx / 400, pcol = idx - r * 400;
    const int gy = pcol / 20, gx = pcol - gy * 20;
    const float s = zwLds[r * 3 + 0], tx = zwLds[r * 3 + 1], ty = zwLds[r * 3 + 2];
    const float bx = (2.0f * gx + 1.0f) * 0.05f - 1.0f;
    const float by = (2.0f * gy + 1.0f) * 0.05f - 1.0f;
    const float ix = ((s * bx + tx + 1.0f) * 50.0f - 1.0f) * 0.5f;
    const float iy = ((s * by + ty + 1.0f) * 50.0f - 1.0f) * 0.5f;
    const float x0f = floorf(ix), y0f = floorf(iy);
    const float wx = ix - x0f, wy = iy - y0f;
    const int x0 = (int)x0f, y0 = (int)y0f;
    const float* im = p.img + (size_t)(row0 + r) * 2500;
    float q[4];
    const int xs[2] = {x0, x0 + 1}, ys[2] = {y0, y0 + 1};
#pragma unroll
    for (int a = 0; a < 2; ++a)
#pragma unroll
      for (int b = 0; b < 2; ++b) {
        const int yi = ys[a], xi = xs[b];
        const bool valid = (xi >= 0) & (xi < 50) & (yi >= 0) & (yi < 50);
        const int yc = min(max(yi, 0), 49), xc = min(max(xi, 0), 49);
        const float vv = im[yc * 50 + xc];
        q[a * 2 + b] = valid ? vv : 0.0f;
      }
    const float val = q[0] * (1 - wx) * (1 - wy) + q[1] * wx * (1 - wy)
                    + q[2] * (1 - wx) * wy       + q[3] * wx * wy;
    U[r * K_GLM + pcol] = f2bf(val);
  }
  __syncthreads();

  // g(16x400) @ w_ge1^T -> relu -> h1(16x200)   [16 tiles, guard-free]
  f8v acc1[2] = {};
  wmma_mm<2>(U, K_GLM, p.W + OFF_WGE1, K_GLM, K_GLM / 32, wave, lane, acc1);
  u16* h1 = U + MT * K_GLM;
  const int rA = lane & 15, hi = lane >> 4;
#pragma unroll
  for (int j = 0; j < 2; ++j) {
    const int col = (wave + 8 * j) * 16 + rA;
    if (col < 200) {
      const float b = p.b_ge1[col];
#pragma unroll
      for (int v = 0; v < 8; ++v)
        h1[(v + hi * 8) * K_GE2 + col] = f2bf(fmaxf(acc1[j][v] + b, 0.0f));
    }
  }
  __syncthreads();

  // h1(16x200) @ w_ge2^T -> enc(16x100)   [8 tiles, guard-free]
  f8v acc2[1] = {};
  wmma_mm<1>(h1, K_GE2, p.W + OFF_WGE2, K_GE2, K_GE2 / 32, wave, lane, acc2);
  {
    const int col = wave * 16 + rA;
    if (col < 100) {
      const float b = p.b_ge2[col];
#pragma unroll
      for (int v = 0; v < 8; ++v)
        encbf[(v + hi * 8) * ENC_W + col] = f2bf(acc2[0][v] + b);
    }
  }
}

__global__ __launch_bounds__(NTHREADS, 1)
void propagate_kernel(KParams p) {
  __shared__ u16 U[10240];              // g / h1 / xA time-aliased (20KB)
  __shared__ u16 hbf[MT * HID];         // h_rela bf16
  __shared__ u16 htbf[MT * HID];        // h_temp bf16
  __shared__ u16 hidbf[MT * HID];       // hidden_last_time_temp bf16
  __shared__ u16 encbf[MT * ENC_W];
  __shared__ float mt_raw[MT * MT_W];
  __shared__ u16 zwtbf[MT * WHAT];      // z_what_last_time (this t)
  __shared__ u16 zwtprevbf[MT * WHAT];  // z_what carry
  __shared__ float zwhat_cur[MT * WHAT];
  __shared__ float zwr[MT * WHERE], zwprev[MT * WHERE], zwcur[MT * WHERE], zwb[MT * WHERE];
  __shared__ float mwhere[MT * WHERE], swhere[MT * WHERE];
  __shared__ float mw6[MT * 6];
  __shared__ float zp[MT], up_s[MT], zpres[MT];
  __shared__ float klh_step[MT], klw_step[MT], klh_acc[MT], klw_acc[MT];

  const int tid = threadIdx.x;
  const int wave = tid >> 5, lane = tid & 31;
  const int rA = lane & 15, hi = lane >> 4;
  const int row0 = blockIdx.x * MT;

  f8v cA = {}, cB = {};   // c_rela carry lives in VGPRs for all 16 steps

  for (int i = tid; i < MT * HID; i += NTHREADS) hbf[i] = 0;
  for (int i = tid; i < MT * WHAT; i += NTHREADS) zwtprevbf[i] = 0;
  if (tid < MT * WHERE) zwprev[tid] = 0.0f;
  if (tid < MT) { klh_acc[tid] = 0.0f; klw_acc[tid] = 0.0f; }
  __syncthreads();

  for (int t = 0; t < TSTEPS; ++t) {
    // ---- per-step inputs -> LDS
    for (int i = tid; i < MT * HID; i += NTHREADS) {
      const int r = i >> 8, d = i & 255;
      hidbf[i] = f2bf(p.hid_last[((size_t)(row0 + r) * TSTEPS + t) * HID + d]);
    }
    for (int i = tid; i < MT * WHAT; i += NTHREADS) {
      const int r = i / WHAT, d = i - r * WHAT;
      zwtbf[i] = f2bf(p.zwt_last[((size_t)(row0 + r) * TSTEPS + t) * WHAT + d]);
    }
    if (tid < MT * WHERE) {
      const int r = tid / WHERE, d = tid - r * WHERE;
      zwr[tid] = p.zwr_last[((size_t)(row0 + r) * TSTEPS + t) * WHERE + d];
    }
    if (tid < MT) {
      zp[tid] = p.zp_last[(size_t)(row0 + tid) * TSTEPS + t];
      up_s[tid] = p.u_pres[(size_t)(row0 + tid) * TSTEPS + t];
    }
    __syncthreads();

    // ---- z_where_bias = relu(hid @ w_pro.T + b) + zwr
    if (tid < MT * WHERE) {
      const int r = tid / WHERE, o = tid - r * WHERE;
      float a = p.b_pro_loca[o];
      for (int k = 0; k < HID; ++k)
        a += bf2f(hidbf[r * HID + k]) * p.w_pro_loca[o * HID + k];
      zwb[tid] = fmaxf(a, 0.0f) + zwr[tid];
    }
    __syncthreads();

    // ---- glimpse encode #1 -> encbf
    glimpse_encode(p, row0, zwb, U, encbf, tid, wave, lane);
    __syncthreads();

    // ---- build rel_in = [enc(100)|zwr(3)|zw_prev(3)|zwt(50)|zwt_prev(50)|hid(256)]
    for (int idx = tid; idx < MT * K_REL; idx += NTHREADS) {
      const int r = idx / K_REL, c = idx - r * K_REL;
      u16 v;
      if      (c < 100) v = encbf[r * ENC_W + c];
      else if (c < 103) v = f2bf(zwr[r * WHERE + (c - 100)]);
      else if (c < 106) v = f2bf(zwprev[r * WHERE + (c - 103)]);
      else if (c < 156) v = zwtbf[r * WHAT + (c - 106)];
      else if (c < 206) v = zwtprevbf[r * WHAT + (c - 156)];
      else if (c < 462) v = hidbf[r * HID + (c - 206)];
      else              v = 0;
      U[r * K_REL + c] = v;
    }
    __syncthreads();

    // ---- rel LSTM gates: x@Wih_rel^T + h@Whh_rel^T   [64 tiles, guard-free]
    f8v acc[8] = {};
    wmma_mm<8>(U, K_REL, p.W + OFF_WIH_REL, K_REL, K_REL / 32, wave, lane, acc);
    wmma_mm<8>(hbf, HID, p.W + OFF_WHH_REL, HID, HID / 32, wave, lane, acc);
    __syncthreads();                               // all waves done reading old hbf
    lstm_epilogue(acc, p.bih_rel, p.bhh_rel, &cA, &cB, true, hbf, wave, lane);
    __syncthreads();

    // ---- mw = [zwr|h_rela] @ w_lwhere^T + b  -> z_where_i
    if (tid < MT * 6) {
      const int r = tid / 6, o = tid - r * 6;
      const float* wrow = p.w_lwhere + o * (HID + WHERE);
      float a = p.b_lwhere[o];
      for (int k = 0; k < WHERE; ++k) a += zwr[r * WHERE + k] * wrow[k];
      for (int k = 0; k < HID; ++k)   a += bf2f(hbf[r * HID + k]) * wrow[WHERE + k];
      mw6[tid] = a;
    }
    __syncthreads();
    if (tid < MT * WHERE) {
      const int r = tid / WHERE, o = tid - r * WHERE;
      const float m = mw6[r * 6 + o];
      const float s = splus(mw6[r * 6 + WHERE + o]) + 1e-4f;
      const float e = p.eps_where[((size_t)(row0 + r) * TSTEPS + t) * WHERE + o];
      mwhere[tid] = m; swhere[tid] = s;
      zwcur[tid] = m + s * e;
    }
    __syncthreads();

    // ---- glimpse encode #2 (z_where_i) -> encbf
    glimpse_encode(p, row0, zwcur, U, encbf, tid, wave, lane);
    __syncthreads();

    // ---- build tem_in = [enc(100)|zw_prev(3)|hid(256)|h_rela(256)]
    for (int idx = tid; idx < MT * K_TEM; idx += NTHREADS) {
      const int r = idx / K_TEM, c = idx - r * K_TEM;
      u16 v;
      if      (c < 100) v = encbf[r * ENC_W + c];
      else if (c < 103) v = f2bf(zwprev[r * WHERE + (c - 100)]);
      else if (c < 359) v = hidbf[r * HID + (c - 103)];
      else if (c < 615) v = hbf[r * HID + (c - 359)];
      else              v = 0;
      U[r * K_TEM + c] = v;
    }
    __syncthreads();

    // ---- tem LSTM gates (reads new h_rela/c_rela, does NOT update c carry)
    f8v acct[8] = {};
    wmma_mm<8>(U, K_TEM, p.W + OFF_WIH_TEM, K_TEM, K_TEM / 32, wave, lane, acct);
    wmma_mm<8>(hbf, HID, p.W + OFF_WHH_TEM, HID, HID / 32, wave, lane, acct);
    __syncthreads();
    lstm_epilogue(acct, p.bih_tem, p.bhh_tem, &cA, &cB, false, htbf, wave, lane);
    __syncthreads();

    // ---- build mt_in = [zwt(50)|h_rela(256)|h_temp(256)]
    for (int idx = tid; idx < MT * K_MT; idx += NTHREADS) {
      const int r = idx / K_MT, c = idx - r * K_MT;
      u16 v;
      if      (c < 50)  v = zwtbf[r * WHAT + c];
      else if (c < 306) v = hbf[r * HID + (c - 50)];
      else if (c < 562) v = htbf[r * HID + (c - 306)];
      else              v = 0;
      U[r * K_MT + c] = v;
    }
    __syncthreads();

    // ---- mt = mt_in @ w_lwhat^T + b   [8 tiles, guard-free]
    f8v accw[1] = {};
    wmma_mm<1>(U, K_MT, p.W + OFF_WLWHAT, K_MT, K_MT / 32, wave, lane, accw);
    {
      const int col = wave * 16 + rA;
      if (col < 100) {
        const float b = p.b_lwhat[col];
#pragma unroll
        for (int v = 0; v < 8; ++v)
          mt_raw[(v + hi * 8) * MT_W + col] = accw[0][v] + b;
      }
    }
    if (tid < MT) { klh_step[tid] = 0.0f; klw_step[tid] = 0.0f; }
    __syncthreads();

    // ---- z_what, KL(what) and KL(where) partial sums
    for (int idx = tid; idx < MT * WHAT; idx += NTHREADS) {
      const int r = idx / WHAT, j2 = idx - r * WHAT;
      const float m = mt_raw[r * MT_W + j2];
      const float s = splus(mt_raw[r * MT_W + WHAT + j2]) + 1e-4f;
      const float e = p.eps_what[((size_t)(row0 + r) * TSTEPS + t) * WHAT + j2];
      zwhat_cur[idx] = m + s * e;
      const float kl = -logf(s) + 0.5f * (s * s + m * m) - 0.5f;
      atomicAdd(&klh_step[r], kl);
    }
    if (tid < MT * WHERE) {
      const int r = tid / WHERE, o = tid - r * WHERE;
      const float m = mwhere[tid], s = swhere[tid];
      const float pm = (o == 0) ? 0.3f : 0.0f;
      const float ps = (o == 0) ? 0.1f : 1.0f;
      const float kl = logf(ps / s) + (s * s + (m - pm) * (m - pm)) / (2.0f * ps * ps) - 0.5f;
      atomicAdd(&klw_step[r], kl);
    }
    __syncthreads();

    // ---- presence head + KL accumulation
    if (tid < MT) {
      const int r = tid;
      const float* wl = p.w_lpres;
      float a = p.b_lpres[0];
      for (int j = 0; j < WHAT; ++j)  a += zwhat_cur[r * WHAT + j] * wl[j];
      for (int o = 0; o < WHERE; ++o) a += zwcur[r * WHERE + o] * wl[WHAT + o];
      for (int k = 0; k < HID; ++k)   a += bf2f(hbf[r * HID + k]) * wl[WHAT + WHERE + k];
      for (int k = 0; k < HID; ++k)   a += bf2f(htbf[r * HID + k]) * wl[WHAT + WHERE + HID + k];
      const float pr = sigf(a);
      const float w = (up_s[r] < pr * zp[r]) ? 1.0f : 0.0f;
      zpres[r] = w;
      klh_acc[r] += klh_step[r] * w;
      klw_acc[r] += klw_step[r] * w;
    }
    __syncthreads();

    // ---- outputs + carry updates
    for (int idx = tid; idx < MT * WHAT; idx += NTHREADS) {
      const int r = idx / WHAT, j2 = idx - r * WHAT;
      const float z = zwhat_cur[idx];
      p.out[OUT_ZWHAT + ((size_t)(row0 + r) * TSTEPS + t) * WHAT + j2] = z;
      zwtprevbf[idx] = f2bf(z);
    }
    if (tid < MT * WHERE) {
      const int r = tid / WHERE, o = tid - r * WHERE;
      p.out[OUT_ZWHERE + ((size_t)(row0 + r) * TSTEPS + t) * WHERE + o] = zwcur[tid];
      zwprev[tid] = zwcur[tid];
    }
    if (tid < MT)
      p.out[OUT_ZPRES + (size_t)(row0 + tid) * TSTEPS + t] = zpres[tid];
    for (int idx = tid; idx < MT * HID; idx += NTHREADS) {
      const int r = idx >> 8, d = idx & 255;
      p.out[OUT_HTEMP + ((size_t)(row0 + r) * TSTEPS + t) * HID + d] = bf2f(htbf[idx]);
    }
    __syncthreads();
  }

  if (tid < MT) {
    p.out[OUT_KLWHAT + row0 + tid] = klh_acc[tid];
    p.out[OUT_KLWHERE + row0 + tid] = klw_acc[tid];
  }
}

// f32 -> zero-padded bf16 row-major weight conversion
__global__ void convert_pad_kernel(const float* __restrict__ src, int rows, int cols,
                                   u16* __restrict__ dst, int prows, int pcols) {
  const int total = prows * pcols;
  for (int i = blockIdx.x * blockDim.x + threadIdx.x; i < total;
       i += gridDim.x * blockDim.x) {
    const int r = i / pcols, c = i - r * pcols;
    const float v = (r < rows && c < cols) ? src[r * cols + c] : 0.0f;
    dst[i] = f2bf(v);
  }
}

extern "C" void kernel_launch(void* const* d_in, const int* in_sizes, int n_in,
                              void* d_out, int out_size, void* d_ws, size_t ws_size,
                              hipStream_t stream) {
  (void)in_sizes; (void)n_in; (void)out_size; (void)ws_size;
  u16* W = (u16*)d_ws;
  auto conv = [&](const void* src, int rows, int cols, unsigned off, int prows, int pcols) {
    const int total = prows * pcols;
    const int blocks = (total + 255) / 256;
    convert_pad_kernel<<<blocks, 256, 0, stream>>>((const float*)src, rows, cols,
                                                   W + off, prows, pcols);
  };
  conv(d_in[14], 1024, 462, OFF_WIH_REL, 1024, K_REL);
  conv(d_in[15], 1024, 256, OFF_WHH_REL, 1024, 256);
  conv(d_in[18], 1024, 615, OFF_WIH_TEM, 1024, K_TEM);
  conv(d_in[19], 1024, 256, OFF_WHH_TEM, 1024, 256);
  conv(d_in[10], 200, 400, OFF_WGE1, N_GE1P, K_GLM);
  conv(d_in[12], 100, 200, OFF_WGE2, N_GE2P, K_GE2);
  conv(d_in[24], 100, 562, OFF_WLWHAT, N_LWHATP, K_MT);

  KParams p;
  p.img        = (const float*)d_in[0];
  p.zwt_last   = (const float*)d_in[1];
  p.zwr_last   = (const float*)d_in[2];
  p.zp_last    = (const float*)d_in[3];
  p.hid_last   = (const float*)d_in[4];
  p.eps_where  = (const float*)d_in[5];
  p.eps_what   = (const float*)d_in[6];
  p.u_pres     = (const float*)d_in[7];
  p.w_pro_loca = (const float*)d_in[8];
  p.b_pro_loca = (const float*)d_in[9];
  p.b_ge1      = (const float*)d_in[11];
  p.b_ge2      = (const float*)d_in[13];
  p.bih_rel    = (const float*)d_in[16];
  p.bhh_rel    = (const float*)d_in[17];
  p.bih_tem    = (const float*)d_in[20];
  p.bhh_tem    = (const float*)d_in[21];
  p.w_lwhere   = (const float*)d_in[22];
  p.b_lwhere   = (const float*)d_in[23];
  p.b_lwhat    = (const float*)d_in[25];
  p.w_lpres    = (const float*)d_in[26];
  p.b_lpres    = (const float*)d_in[27];
  p.W = W;
  p.out = (float*)d_out;

  propagate_kernel<<<NROW / MT, NTHREADS, 0, stream>>>(p);
}
